// BiMambaTextEncoder_52913997086991
// MI455X (gfx1250) — matrix operand order repaired and checked by reading, not compiled, not run on hardware
//
#include <hip/hip_runtime.h>

// ---------------------------------------------------------------------------
// BiMamba text encoder for MI455X (gfx1250, wave32, WMMA bf16, async-LDS)
// B=8 L=512 C=512  DI=1024 DS=16 DTR=32 DC=4  K=5 DEPTH=3
// ---------------------------------------------------------------------------

#define BB 8
#define LL 512
#define CC 512
#define DI 1024
#define DS 16
#define DTR 32
#define ROWS (BB*LL)          // 4096

typedef __attribute__((ext_vector_type(16))) __bf16 v16bf;
typedef __attribute__((ext_vector_type(8)))  float  v8f;

union BF16Frag { v16bf v; unsigned int u[8]; };
union AccFrag  { v8f   v; float        f[8]; };

// ---- gfx1250 async global->LDS copy path (guarded; falls back to ds_store) --
#if defined(__has_builtin)
#if __has_builtin(__builtin_amdgcn_global_load_async_to_lds_b128) && \
    __has_builtin(__builtin_amdgcn_s_wait_asynccnt)
#define USE_ASYNC_LDS 1
#endif
#endif
#ifndef USE_ASYNC_LDS
#define USE_ASYNC_LDS 0
#endif

#if USE_ASYNC_LDS
// builtin parameter type (from clang diagnostic): pointer to
//   __attribute__((__vector_size__(4*sizeof(int)))) int
// in AS(1) (printed "__device__") for global, AS(3) for LDS.
typedef __attribute__((__vector_size__(4 * sizeof(int)))) int vs4i;
// flat global address == as(1) address; low 32 bits of flat LDS address == LDS offset
#define TO_GLB128(p) ((__attribute__((address_space(1))) vs4i*)(unsigned long long)(const void*)(p))
#define TO_LDS128(p) ((__attribute__((address_space(3))) vs4i*)(unsigned int)(unsigned long long)(const void*)(p))
#endif

__device__ __forceinline__ unsigned short f2bf(float f) {
  unsigned int u = __float_as_uint(f);
  u += 0x7FFFu + ((u >> 16) & 1u);
  return (unsigned short)(u >> 16);
}
__device__ __forceinline__ float sigmoidf_(float x) { return 1.0f / (1.0f + __expf(-x)); }
__device__ __forceinline__ float softplusf_(float x) { return x > 20.0f ? x : log1pf(__expf(x)); }

// ---------------------------------------------------------------------------
// Generic bf16 WMMA GEMM:  Y[M,N](f32) = A[M,K](bf16, lda) @ W[N,K](bf16, ldw)^T
// mode: 0 = none, 1 = +bias, 2 = softplus(+bias)
// block 256 threads (8 waves), tile 128x64, K-step 32, 4 wmma / wave / step
// ---------------------------------------------------------------------------
__global__ __launch_bounds__(256)
void wmma_gemm_kernel(const unsigned short* __restrict__ A, int lda,
                      const unsigned short* __restrict__ W, int ldw,
                      float* __restrict__ Y, int ldy,
                      const float* __restrict__ bias,
                      int M, int N, int Kdim, int mode)
{
  __shared__ unsigned short As[128][32];
  __shared__ unsigned short Bs[64][32];

  const int tid  = threadIdx.x;
  const int lane = tid & 31, wave = tid >> 5;
  const int wm = wave & 3, wn = wave >> 2;
  const int m0 = blockIdx.y * 128, n0 = blockIdx.x * 64;
  const int hi = lane >> 4, lo = lane & 15;

  AccFrag acc[2][2];
#pragma unroll
  for (int i = 0; i < 2; i++)
#pragma unroll
    for (int j = 0; j < 2; j++)
#pragma unroll
      for (int v = 0; v < 8; v++) acc[i][j].f[v] = 0.0f;

  const int ar = tid >> 1, ac = (tid & 1) * 16;   // A cooperative load: 2 thr/row
  const int br = tid >> 2, bc = (tid & 3) * 8;    // W cooperative load: 4 thr/row

  for (int k0 = 0; k0 < Kdim; k0 += 32) {
    const unsigned short* agp = A + (size_t)(m0 + ar) * lda + k0 + ac;
    const unsigned short* bgp = W + (size_t)(n0 + br) * ldw + k0 + bc;
#if USE_ASYNC_LDS
    // per-lane 16B async copies, LDS write bypasses VGPRs; tracked by ASYNCcnt
    __builtin_amdgcn_global_load_async_to_lds_b128(TO_GLB128(agp), TO_LDS128(&As[ar][ac]), 0, 0);
    __builtin_amdgcn_global_load_async_to_lds_b128(TO_GLB128(agp), TO_LDS128(&As[ar][ac]), 16, 0);
    __builtin_amdgcn_global_load_async_to_lds_b128(TO_GLB128(bgp), TO_LDS128(&Bs[br][bc]), 0, 0);
    if (k0 + 32 < Kdim) {
      __builtin_prefetch(agp + 32, 0, 1);
      __builtin_prefetch(bgp + 32, 0, 1);
    }
    __builtin_amdgcn_s_wait_asynccnt(0);
    __syncthreads();
#else
    const uint4* ap = (const uint4*)agp;
    *(uint4*)&As[ar][ac]     = ap[0];
    *(uint4*)&As[ar][ac + 8] = ap[1];
    *(uint4*)&Bs[br][bc]     = *(const uint4*)bgp;
    if (k0 + 32 < Kdim) {
      __builtin_prefetch(agp + 32, 0, 1);
      __builtin_prefetch(bgp + 32, 0, 1);
    }
    __syncthreads();
#endif

    BF16Frag afrag[2], bfrag[2];
#pragma unroll
    for (int i = 0; i < 2; i++) {
      const int mrow = wm * 32 + i * 16 + lo;
#pragma unroll
      for (int v = 0; v < 8; v++) {
        const int kk = 2 * (v & 3) + 8 * hi + 16 * (v >> 2);   // ISA 16-bit A 16x32 layout
        afrag[i].u[v] = *(const unsigned int*)&As[mrow][kk];
      }
    }
#pragma unroll
    for (int j = 0; j < 2; j++) {
      const int ncol = wn * 32 + j * 16 + lo;
#pragma unroll
      for (int v = 0; v < 8; v++) {
        const int kk = 2 * (v & 3) + 8 * hi + 16 * (v >> 2);   // ISA 16-bit B 32x16 layout (same k order as A)
        bfrag[j].u[v] = *(const unsigned int*)&Bs[ncol][kk];
      }
    }
#pragma unroll
    for (int i = 0; i < 2; i++)
#pragma unroll
      for (int j = 0; j < 2; j++)
        acc[i][j].v = __builtin_amdgcn_wmma_f32_16x16x32_bf16(
            false, afrag[i].v, false, bfrag[j].v, (short)0, acc[i][j].v, false, false);
    __syncthreads();
  }

#pragma unroll
  for (int i = 0; i < 2; i++) {
#pragma unroll
    for (int j = 0; j < 2; j++) {
      const int col = n0 + wn * 32 + j * 16 + lo;
      const float bv = (mode >= 1 && bias) ? bias[col] : 0.0f;
#pragma unroll
      for (int v = 0; v < 8; v++) {
        const int row = m0 + wm * 32 + i * 16 + v + 8 * hi;    // ISA C/D layout
        float val = acc[i][j].f[v] + bv;
        if (mode == 2) val = softplusf_(val);
        Y[(size_t)row * ldy + col] = val;
      }
    }
  }
}

// ---------------------------------------------------------------------------
// Embedding gather -> (B,C,L) fp32
// ---------------------------------------------------------------------------
__global__ void embed_kernel(const int* __restrict__ tok,
                             const float* __restrict__ emb,
                             float* __restrict__ hcl)
{
  int idx = blockIdx.x * blockDim.x + threadIdx.x;   // (b*C + c)*L + l
  if (idx >= BB * CC * LL) return;
  int l = idx & (LL - 1);
  int c = (idx >> 9) & (CC - 1);
  int b = idx >> 18;
  hcl[idx] = emb[(size_t)tok[b * LL + l] * CC + c];
}

// ---------------------------------------------------------------------------
// Grouped conv1d (groups=128, K=5, same pad) + GroupNorm(4) + leaky(0.2) + mask
// one block per (batch, gn-group): stats over 128 channels x 512 L.
// Pass 1: conv -> scratch (+ local sum/sumsq), reduce; Pass 2: normalize.
// last!=0: write (B,L,C) bf16 forward + L-reversed copies.
// ---------------------------------------------------------------------------
__global__ __launch_bounds__(256)
void convgn_kernel(const float* __restrict__ in,    // (B,C,L)
                   float* __restrict__ scratch,     // (B,C,L)
                   const float* __restrict__ w,     // (C,4,5)
                   const float* __restrict__ bconv,
                   const float* __restrict__ gs,
                   const float* __restrict__ gb,
                   const unsigned char* __restrict__ mask,
                   int last,
                   unsigned short* __restrict__ hbf,
                   unsigned short* __restrict__ hbfrev)
{
  __shared__ float r1[8], r2[8], stats[2];
  const int b = blockIdx.x >> 2;
  const int g = blockIdx.x & 3;
  const int tid = threadIdx.x;
  float s1 = 0.0f, s2 = 0.0f;

  for (int i = tid; i < 128 * LL; i += 256) {
    const int cl = i >> 9, l = i & (LL - 1);
    const int c = g * 128 + cl;
    const int cin0 = c & ~3;
    float acc = bconv[c];
#pragma unroll
    for (int ci = 0; ci < 4; ci++) {
      const float* ip = in + ((size_t)b * CC + cin0 + ci) * LL;
      const float* wp = w + (size_t)(c * 4 + ci) * 5;
#pragma unroll
      for (int k = 0; k < 5; k++) {
        const int lsrc = l + k - 2;
        if (lsrc >= 0 && lsrc < LL) acc += wp[k] * ip[lsrc];
      }
    }
    scratch[((size_t)b * CC + c) * LL + l] = acc;
    s1 += acc; s2 += acc * acc;
  }
#pragma unroll
  for (int o = 16; o > 0; o >>= 1) { s1 += __shfl_xor(s1, o, 32); s2 += __shfl_xor(s2, o, 32); }
  if ((tid & 31) == 0) { r1[tid >> 5] = s1; r2[tid >> 5] = s2; }
  __syncthreads();
  if (tid == 0) {
    float a = 0.0f, q = 0.0f;
    for (int i = 0; i < 8; i++) { a += r1[i]; q += r2[i]; }
    const float mu = a / 65536.0f;
    stats[0] = mu;
    stats[1] = rsqrtf(q / 65536.0f - mu * mu + 1e-5f);
  }
  __syncthreads();
  const float mu = stats[0], rstd = stats[1];

  for (int i = tid; i < 128 * LL; i += 256) {
    const int cl = i >> 9, l = i & (LL - 1);
    const int c = g * 128 + cl;
    const size_t p = ((size_t)b * CC + c) * LL + l;
    float v = (scratch[p] - mu) * rstd * gs[c] + gb[c];
    v = v < 0.0f ? 0.2f * v : v;
    if (mask[b * LL + l]) v = 0.0f;
    if (last) {
      const unsigned short h = f2bf(v);
      hbf   [((size_t)b * LL + l)            * CC + c] = h;
      hbfrev[((size_t)b * LL + (LL - 1 - l)) * CC + c] = h;
    } else {
      scratch[p] = v;
    }
  }
}

// ---------------------------------------------------------------------------
// Causal depthwise conv (DC=4, left pad 3) + SiLU. xi_pre = xz cols [0,DI)
// ---------------------------------------------------------------------------
__global__ void dwconv_silu_kernel(const float* __restrict__ xz,
                                   const float* __restrict__ cw,  // (DI,1,4)
                                   const float* __restrict__ cb,
                                   float* __restrict__ xi32,
                                   unsigned short* __restrict__ xibf)
{
  const int idx = blockIdx.x * blockDim.x + threadIdx.x;  // bl*DI + d
  if (idx >= ROWS * DI) return;
  const int d  = idx & (DI - 1);
  const int bl = idx >> 10;
  const int l  = bl & (LL - 1);
  float acc = cb[d];
#pragma unroll
  for (int k = 0; k < 4; k++) {
    const int lsrc = l + k - 3;
    if (lsrc >= 0) acc += cw[d * 4 + k] * xz[(size_t)(bl - l + lsrc) * (2 * DI) + d];
  }
  const float s = acc * sigmoidf_(acc);
  xi32[idx] = s;
  xibf[idx] = f2bf(s);
}

// ---------------------------------------------------------------------------
// Selective scan. block = (b, d-chunk of 256). ybf = (y + xi*D) * silu(z)  (bf16)
// ---------------------------------------------------------------------------
__global__ __launch_bounds__(256)
void scan_kernel(const float* __restrict__ xz,     // z at col DI+d
                 const float* __restrict__ xi32,
                 const float* __restrict__ dbl,    // (rows,64): dt|B|C
                 const float* __restrict__ delta,
                 const float* __restrict__ A_log,  // (DI,16)
                 const float* __restrict__ Dp,
                 unsigned short* __restrict__ ybf)
{
  __shared__ float bc[32];                         // Bm[16] | Cm[16]
  const int b = blockIdx.x >> 2;
  const int d = ((blockIdx.x & 3) << 8) | threadIdx.x;
  float Av[DS], h[DS];
#pragma unroll
  for (int s = 0; s < DS; s++) { Av[s] = -__expf(A_log[d * DS + s]); h[s] = 0.0f; }
  const float Dd = Dp[d];

  for (int l = 0; l < LL; l++) {
    const size_t row = (size_t)b * LL + l;
    if (threadIdx.x < 32) bc[threadIdx.x] = dbl[row * 64 + 32 + threadIdx.x];
    __syncthreads();
    const float dl = delta[row * DI + d];
    const float xv = xi32[row * DI + d];
    const float zv = xz[row * (2 * DI) + DI + d];
    float y = 0.0f;
#pragma unroll
    for (int s = 0; s < DS; s++) {
      const float da = __expf(dl * Av[s]);
      h[s] = da * h[s] + dl * bc[s] * xv;
      y += h[s] * bc[16 + s];
    }
    y += xv * Dd;
    ybf[row * DI + d] = f2bf(y * (zv * sigmoidf_(zv)));
    __syncthreads();
  }
}

// ---------------------------------------------------------------------------
// Fuse epilogue: val = cross + silu(g)*combined; LayerNorm(1024); -> bf16
// ---------------------------------------------------------------------------
__global__ __launch_bounds__(256)
void fuse_ln_kernel(const float* __restrict__ cross,
                    const float* __restrict__ g,
                    const float* __restrict__ fo,
                    const float* __restrict__ bo,
                    const float* __restrict__ ln_s,
                    const float* __restrict__ ln_b,
                    unsigned short* __restrict__ outbf)
{
  __shared__ float buf[2 * CC];
  __shared__ float r1[8], r2[8], st[2];
  const int row = blockIdx.x, tid = threadIdx.x;
  float s1 = 0.0f, s2 = 0.0f;
  for (int j = tid; j < 2 * CC; j += 256) {
    const float comb = (j < CC) ? fo[(size_t)row * CC + j] : bo[(size_t)row * CC + j - CC];
    const float gv = g[(size_t)row * 2 * CC + j];
    const float val = cross[(size_t)row * 2 * CC + j] + comb * (gv * sigmoidf_(gv));
    buf[j] = val; s1 += val; s2 += val * val;
  }
#pragma unroll
  for (int o = 16; o > 0; o >>= 1) { s1 += __shfl_xor(s1, o, 32); s2 += __shfl_xor(s2, o, 32); }
  if ((tid & 31) == 0) { r1[tid >> 5] = s1; r2[tid >> 5] = s2; }
  __syncthreads();
  if (tid == 0) {
    float a = 0.0f, q = 0.0f;
    for (int i = 0; i < 8; i++) { a += r1[i]; q += r2[i]; }
    const float mu = a / 1024.0f;
    st[0] = mu; st[1] = rsqrtf(q / 1024.0f - mu * mu + 1e-5f);
  }
  __syncthreads();
  const float mu = st[0], rstd = st[1];
  for (int j = tid; j < 2 * CC; j += 256)
    outbf[(size_t)row * 2 * CC + j] = f2bf((buf[j] - mu) * rstd * ln_s[j] + ln_b[j]);
}

// ---------------------------------------------------------------------------
// small elementwise helpers
// ---------------------------------------------------------------------------
__global__ void f2bf_kernel(const float* __restrict__ s, unsigned short* __restrict__ d, int n) {
  int i = blockIdx.x * blockDim.x + threadIdx.x;
  if (i < n) d[i] = f2bf(s[i]);
}
__global__ void dtbf_kernel(const float* __restrict__ dbl, unsigned short* __restrict__ dt) {
  int i = blockIdx.x * blockDim.x + threadIdx.x;        // row*32 + j
  if (i >= ROWS * DTR) return;
  dt[i] = f2bf(dbl[(size_t)(i >> 5) * 64 + (i & 31)]);
}
__global__ void combbf_kernel(const float* __restrict__ fo, const float* __restrict__ bo,
                              unsigned short* __restrict__ cb) {
  int i = blockIdx.x * blockDim.x + threadIdx.x;        // row*1024 + j
  if (i >= ROWS * 2 * CC) return;
  int j = i & (2 * CC - 1), row = i >> 10;
  cb[i] = f2bf(j < CC ? fo[(size_t)row * CC + j] : bo[(size_t)row * CC + j - CC]);
}
__global__ void maskout_kernel(const float* __restrict__ y, const unsigned char* __restrict__ m,
                               float* __restrict__ out) {
  int i = blockIdx.x * blockDim.x + threadIdx.x;        // row*512 + c
  if (i >= ROWS * CC) return;
  out[i] = m[i >> 9] ? 0.0f : y[i];
}

// ---------------------------------------------------------------------------
// host side
// ---------------------------------------------------------------------------
struct MambaP {
  const float *in_proj, *conv_w, *conv_b, *x_proj, *dt_w, *dt_b, *A_log, *D, *out_proj;
};
struct CnnP { const float *w, *b, *gs, *gb; };

extern "C" void kernel_launch(void* const* d_in, const int* in_sizes, int n_in,
                              void* d_out, int out_size, void* d_ws, size_t ws_size,
                              hipStream_t stream)
{
  if (n_in < 43) return;
  auto F = [&](int i) -> const float* { return (const float*)d_in[i]; };

  // ---- input index mapping (robust to insertion-order vs sorted-key flattening)
  const bool topIns = (in_sizes[0] == BB * LL && in_sizes[1] == BB * 128);
  int xI, mI, embI, cnnB, mfB, mbB, fuseB;
  if (topIns) { xI = 0; mI = 3; embI = 4; cnnB = 5; mfB = 17; mbB = 26; fuseB = 35; }
  else        { cnnB = 0; embI = 12; fuseB = 13; mI = 22; mbB = 23; mfB = 32; xI = 42; }

  const int* tok = (const int*)d_in[xI];
  const unsigned char* mptr = (const unsigned char*)d_in[mI];
  const float* embp = F(embI);

  CnnP cnn[3];
  const bool cnnIns = (in_sizes[cnnB] == CC * 4 * 5);
  for (int i = 0; i < 3; i++) {
    const int base = cnnB + 4 * i;
    if (cnnIns) { cnn[i].w = F(base); cnn[i].b = F(base + 1); cnn[i].gs = F(base + 2); cnn[i].gb = F(base + 3); }
    else        { cnn[i].b = F(base); cnn[i].gb = F(base + 1); cnn[i].gs = F(base + 2); cnn[i].w = F(base + 3); }
  }

  MambaP mp[2];
  const int mbase[2] = { mfB, mbB };
  for (int d = 0; d < 2; d++) {
    const int base = mbase[d];
    MambaP& p = mp[d];
    if (in_sizes[base] == 2 * DI * CC) {   // insertion order
      p.in_proj = F(base); p.conv_w = F(base + 1); p.conv_b = F(base + 2);
      p.x_proj = F(base + 3); p.dt_w = F(base + 4); p.dt_b = F(base + 5);
      p.A_log = F(base + 6); p.D = F(base + 7); p.out_proj = F(base + 8);
    } else {                               // sorted keys
      p.A_log = F(base); p.D = F(base + 1); p.conv_b = F(base + 2); p.conv_w = F(base + 3);
      p.dt_b = F(base + 4); p.dt_w = F(base + 5); p.in_proj = F(base + 6);
      p.out_proj = F(base + 7); p.x_proj = F(base + 8);
    }
  }

  const float *cross_w, *gw1, *gb1, *gw2, *gb2, *lns, *lnb, *projw;
  if (in_sizes[fuseB + 1] == CC * CC) {    // insertion order
    cross_w = F(fuseB); gw1 = F(fuseB + 1); gb1 = F(fuseB + 2); gw2 = F(fuseB + 3);
    gb2 = F(fuseB + 4); lns = F(fuseB + 5); lnb = F(fuseB + 6); projw = F(fuseB + 7);
  } else {                                 // sorted keys
    cross_w = F(fuseB); gb1 = F(fuseB + 1); gb2 = F(fuseB + 2); gw1 = F(fuseB + 3);
    gw2 = F(fuseB + 4); lnb = F(fuseB + 5); lns = F(fuseB + 6); projw = F(fuseB + 7);
  }

  // ---- workspace layout
  char* ws = (char*)d_ws;
  size_t off = 0;
  auto alloc = [&](size_t bytes) -> char* {
    char* p = ws + off;
    off = (off + bytes + 255) & ~(size_t)255;
    return p;
  };
  float* hcl_a = (float*)alloc((size_t)BB * CC * LL * 4);       // reused as fo
  float* hcl_b = (float*)alloc((size_t)BB * CC * LL * 4);       // reused as bo
  unsigned short* hbf    = (unsigned short*)alloc((size_t)ROWS * CC * 2);
  unsigned short* hbfrev = (unsigned short*)alloc((size_t)ROWS * CC * 2);
  unsigned short* w_in[2]  = { (unsigned short*)alloc(2 * DI * CC * 2), (unsigned short*)alloc(2 * DI * CC * 2) };
  unsigned short* w_xp[2]  = { (unsigned short*)alloc(64 * DI * 2),     (unsigned short*)alloc(64 * DI * 2) };
  unsigned short* w_dt[2]  = { (unsigned short*)alloc(DI * DTR * 2),    (unsigned short*)alloc(DI * DTR * 2) };
  unsigned short* w_op[2]  = { (unsigned short*)alloc(CC * DI * 2),     (unsigned short*)alloc(CC * DI * 2) };
  unsigned short* w_cross  = (unsigned short*)alloc(2 * CC * 2 * CC * 2);
  unsigned short* w_g1     = (unsigned short*)alloc(CC * CC * 2);
  unsigned short* w_g2     = (unsigned short*)alloc(CC * CC * 2);
  unsigned short* w_proj   = (unsigned short*)alloc(CC * 2 * CC * 2);

  // scratch region: mamba layout, later aliased by fuse layout
  char* scr = alloc(86 * 1024 * 1024);
  float* xz            = (float*)(scr);                               // rows x 2048 f32
  float* xi32          = (float*)(scr + 33554432);                    // rows x 1024 f32
  unsigned short* xibf = (unsigned short*)(scr + 50331648);           // rows x 1024 bf16
  float* dblb          = (float*)(scr + 58720256);                    // rows x 64 f32
  unsigned short* dtbf = (unsigned short*)(scr + 59768832);           // rows x 32 bf16
  float* delta         = (float*)(scr + 60031232);                    // rows x 1024 f32
  unsigned short* ybf  = (unsigned short*)(scr + 76808448);           // rows x 1024 bf16
  // fuse aliases (used only after both mamba directions complete)
  unsigned short* combbf   = (unsigned short*)(scr);                  // rows x 1024 bf16
  float* gbuf              = (float*)(scr + 8388608);                 // rows x 1024 f32
  float* crossb            = (float*)(scr + 25165824);                // rows x 1024 f32
  unsigned short* comblnbf = (unsigned short*)(scr + 41943040);       // rows x 1024 bf16
  float* outbuf            = (float*)(scr + 50331648);                // rows x 512 f32
  (void)ws_size;

  // ---- weight f32 -> bf16
  auto cw = [&](const float* s, unsigned short* d, int n) {
    f2bf_kernel<<<(n + 255) / 256, 256, 0, stream>>>(s, d, n);
  };
  for (int d = 0; d < 2; d++) {
    cw(mp[d].in_proj,  w_in[d], 2 * DI * CC);
    cw(mp[d].x_proj,   w_xp[d], 64 * DI);
    cw(mp[d].dt_w,     w_dt[d], DI * DTR);
    cw(mp[d].out_proj, w_op[d], CC * DI);
  }
  cw(cross_w, w_cross, 2 * CC * 2 * CC);
  cw(gw1, w_g1, CC * CC);
  cw(gw2, w_g2, CC * CC);
  cw(projw, w_proj, CC * 2 * CC);

  // ---- embedding + CNN stack
  embed_kernel<<<(BB * CC * LL) / 256, 256, 0, stream>>>(tok, embp, hcl_a);
  convgn_kernel<<<32, 256, 0, stream>>>(hcl_a, hcl_b, cnn[0].w, cnn[0].b, cnn[0].gs, cnn[0].gb,
                                        mptr, 0, nullptr, nullptr);
  convgn_kernel<<<32, 256, 0, stream>>>(hcl_b, hcl_a, cnn[1].w, cnn[1].b, cnn[1].gs, cnn[1].gb,
                                        mptr, 0, nullptr, nullptr);
  convgn_kernel<<<32, 256, 0, stream>>>(hcl_a, hcl_b, cnn[2].w, cnn[2].b, cnn[2].gs, cnn[2].gb,
                                        mptr, 1, hbf, hbfrev);

  float* fo = hcl_a;   // conv buffers are dead now; reuse as mamba outputs
  float* bo = hcl_b;

  // ---- two mamba directions (backward = same pipeline on L-reversed input)
  for (int dir = 0; dir < 2; dir++) {
    const unsigned short* Abf = dir ? hbfrev : hbf;
    // in_proj: (4096x512) @ (2048x512)^T -> xz (4096x2048)
    wmma_gemm_kernel<<<dim3(2 * DI / 64, ROWS / 128), 256, 0, stream>>>(
        Abf, CC, w_in[dir], CC, xz, 2 * DI, nullptr, ROWS, 2 * DI, CC, 0);
    // causal depthwise conv + silu
    dwconv_silu_kernel<<<(ROWS * DI) / 256, 256, 0, stream>>>(
        xz, mp[dir].conv_w, mp[dir].conv_b, xi32, xibf);
    // x_proj: (4096x1024) @ (64x1024)^T -> dbl (4096x64)
    wmma_gemm_kernel<<<dim3(1, ROWS / 128), 256, 0, stream>>>(
        xibf, DI, w_xp[dir], DI, dblb, 64, nullptr, ROWS, 64, DI, 0);
    // dt slice -> bf16
    dtbf_kernel<<<(ROWS * DTR) / 256, 256, 0, stream>>>(dblb, dtbf);
    // dt_proj + bias + softplus: (4096x32) @ (1024x32)^T -> delta
    wmma_gemm_kernel<<<dim3(DI / 64, ROWS / 128), 256, 0, stream>>>(
        dtbf, DTR, w_dt[dir], DTR, delta, DI, mp[dir].dt_b, ROWS, DI, DTR, 2);
    // selective scan -> ybf = (y + xi*D)*silu(z)
    scan_kernel<<<BB * 4, 256, 0, stream>>>(xz, xi32, dblb, delta,
                                            mp[dir].A_log, mp[dir].D, ybf);
    // out_proj: (4096x1024) @ (512x1024)^T -> fo/bo
    wmma_gemm_kernel<<<dim3(CC / 64, ROWS / 128), 256, 0, stream>>>(
        ybf, DI, w_op[dir], DI, dir ? bo : fo, CC, nullptr, ROWS, CC, DI, 0);
  }

  // ---- fuse
  combbf_kernel<<<(ROWS * 2 * CC) / 256, 256, 0, stream>>>(fo, bo, combbf);
  // gates (bias fused)
  wmma_gemm_kernel<<<dim3(CC / 64, ROWS / 128), 256, 0, stream>>>(
      combbf, 2 * CC, w_g1, CC, gbuf, 2 * CC, gb1, ROWS, CC, CC, 1);
  wmma_gemm_kernel<<<dim3(CC / 64, ROWS / 128), 256, 0, stream>>>(
      combbf + CC, 2 * CC, w_g2, CC, gbuf + CC, 2 * CC, gb2, ROWS, CC, CC, 1);
  // cross projection
  wmma_gemm_kernel<<<dim3(2 * CC / 64, ROWS / 128), 256, 0, stream>>>(
      combbf, 2 * CC, w_cross, 2 * CC, crossb, 2 * CC, nullptr, ROWS, 2 * CC, 2 * CC, 0);
  // gated add + layernorm -> bf16
  fuse_ln_kernel<<<ROWS, 256, 0, stream>>>(crossb, gbuf, fo, bo, lns, lnb, comblnbf);
  // final projection
  wmma_gemm_kernel<<<dim3(CC / 64, ROWS / 128), 256, 0, stream>>>(
      comblnbf, 2 * CC, w_proj, 2 * CC, outbuf, CC, nullptr, ROWS, CC, 2 * CC, 0);
  // mask + write output
  maskout_kernel<<<(ROWS * CC) / 256, 256, 0, stream>>>(outbuf, mptr, (float*)d_out);
}
